// NARX_4750233829437
// MI455X (gfx1250) — matrix operand order repaired and checked by reading, not compiled, not run on hardware
//
#include <hip/hip_runtime.h>

typedef _Float16 h8   __attribute__((ext_vector_type(8)));
typedef _Float16 v16h __attribute__((ext_vector_type(16)));
typedef float    v8f  __attribute__((ext_vector_type(8)));

#define D_I   8
#define D_O   8
#define D_X   32
#define D_Y   16
#define D_HL  256
#define T_LEN 512
#define IN_SZ 384     // D_I*D_X + D_O*D_Y
#define BTILE 16
#define KF1   (IN_SZ / 32)   // 12 K-fragments in GEMM1
#define KF2   (D_HL / 32)    // 8  K-fragments in GEMM2

__device__ __forceinline__ v16h cat16(h8 lo, h8 hi) {
  return __builtin_shufflevector(lo, hi, 0,1,2,3,4,5,6,7,8,9,10,11,12,13,14,15);
}

__device__ __forceinline__ v8f wmma_f16(v16h a, v16h b, v8f c) {
  return __builtin_amdgcn_wmma_f32_16x16x32_f16(false, a, false, b, (short)0, c,
                                                false, false);
}

// Convert 16 contiguous f32 (32B-aligned) to a v16h B/A fragment register.
__device__ __forceinline__ v16h cvt16(const float* __restrict__ p) {
  v8f f0 = *(const v8f*)(p);
  v8f f1 = *(const v8f*)(p + 8);
  v16h r;
  #pragma unroll
  for (int i = 0; i < 8; ++i) { r[i] = (_Float16)f0[i]; r[i + 8] = (_Float16)f1[i]; }
  return r;
}

__launch_bounds__(256, 1)
__global__ void narx_persistent(const float* __restrict__ x,
                                const float* __restrict__ W1,
                                const float* __restrict__ b1,
                                const float* __restrict__ W2,
                                const float* __restrict__ b2,
                                float* __restrict__ out)
{
  extern __shared__ __align__(16) char smem[];
  _Float16* sW2  = (_Float16*)smem;            // [16][256]  f16     8192 B
  float*    sB1  = (float*)(sW2 + D_Y * D_HL); // [256]              1024 B
  float*    sB2  = sB1 + D_HL;                 // [16]                 64 B
  _Float16* sInp = (_Float16*)(sB2 + D_Y);     // [16][384]  f16    12288 B
  _Float16* sH   = sInp + BTILE * IN_SZ;       // [16][256]  f16     8192 B
  _Float16* sHist= sH + BTILE * D_HL;          // [8][16][16] f16    4096 B
                                               // total ~33.9 KB

  const int tid  = threadIdx.x;
  const int lane = tid & 31;
  const int wave = tid >> 5;
  const int b0   = blockIdx.x * BTILE;

  const int Nc     = lane & 15;      // column / A-row within 16x16 tile
  const int hiHalf = lane >> 4;      // lane group 0/1
  const int mBase  = hiHalf * 8;     // C/D: M = vgpr_idx + 8*hiHalf
  const int nb0 = wave * 2, nb1 = wave * 2 + 1;   // hidden N-tiles per wave

  // ---- one-time init ----
  for (int i = tid; i < D_Y * D_HL; i += 256) sW2[i] = (_Float16)W2[i];
  sB1[tid] = b1[tid];
  if (tid < D_Y) sB2[tid] = b2[tid];
  for (int i = tid; i < D_O * 16 * D_Y; i += 256) sHist[i] = (_Float16)0.0f;
  for (int i = tid; i < BTILE * D_I * D_Y; i += 256) {
    int m = i / (D_I * D_Y);
    int r = i % (D_I * D_Y);
    out[((size_t)(b0 + m) * T_LEN) * D_Y + r] = 0.0f;
  }

  // ---- hoist W1 B-fragments into registers (loop-invariant over t) ----
  // B[k][n] = W1[n][k]; lane<16: col N=Nc, K kf*32+0..15; lane>=16: K kf*32+16..31
  v16h w1b0[KF1], w1b1[KF1];
  #pragma unroll
  for (int kf = 0; kf < KF1; ++kf) {
    w1b0[kf] = cvt16(W1 + (nb0 * 16 + Nc) * IN_SZ + kf * 32 + hiHalf * 16);
    w1b1[kf] = cvt16(W1 + (nb1 * 16 + Nc) * IN_SZ + kf * 32 + hiHalf * 16);
  }
  __syncthreads();

  const float bias0 = sB1[nb0 * 16 + Nc];
  const float bias1 = sB1[nb1 * 16 + Nc];
  const float bias2 = sB2[Nc];

  // staging decomposition: thread (m = tid>>4, q = tid&15) owns 16 contiguous
  // x floats of row m: time offset dt = q>>1, col base (q&1)*16.
  const int sm  = tid >> 4;
  const int sq  = tid & 15;
  const int sdt = sq >> 1;
  const int skx = (sq & 1) * 16;

  for (int t = D_I; t < T_LEN; ++t) {
    // ---- stage inp = [x window (256) | history (128)] as f16 [16][384] ----
    {
      const float* xp =
          x + (((size_t)(b0 + sm)) * T_LEN + (t - D_I + sdt)) * D_X + skx;
      v16h xv = cvt16(xp);
      h8* dst = (h8*)(sInp + sm * IN_SZ + sq * 16);
      dst[0] = __builtin_shufflevector(xv, xv, 0, 1, 2, 3, 4, 5, 6, 7);
      dst[1] = __builtin_shufflevector(xv, xv, 8, 9, 10, 11, 12, 13, 14, 15);
      if (tid < 128) {              // history: (m, o) pair -> 16 halves
        int hm = tid >> 3, ho = tid & 7;           // o-th oldest = time t-8+o
        const h8* src = (const h8*)(sHist + ((((t + ho) & 7) * 16 + hm) << 4));
        h8* hdst = (h8*)(sInp + hm * IN_SZ + 256 + ho * 16);
        hdst[0] = src[0];
        hdst[1] = src[1];
      }
    }
    __syncthreads();

    // ---- GEMM1: h[16,256] = tanh(inp[16,384] @ W1^T + b1), 2 N-tiles/wave ----
    v8f acc0, acc1;
    #pragma unroll
    for (int i = 0; i < 8; ++i) { acc0[i] = bias0; acc1[i] = bias1; }

    #pragma unroll
    for (int kf = 0; kf < KF1; ++kf) {
      // A fragment: row M = Nc; lane<16 -> K 0..15, lane>=16 -> K 16..31
      const _Float16* ap = sInp + Nc * IN_SZ + kf * 32 + hiHalf * 16;
      v16h a = cat16(*(const h8*)(ap), *(const h8*)(ap + 8));
      acc0 = wmma_f16(a, w1b0[kf], acc0);
      acc1 = wmma_f16(a, w1b1[kf], acc1);
    }

    #pragma unroll
    for (int r = 0; r < 8; ++r) {
      // tanh(v) = 1 - 2/(exp(2v)+1)
      float e0 = __expf(2.0f * acc0[r]);
      float e1 = __expf(2.0f * acc1[r]);
      int M = mBase + r;
      sH[M * D_HL + nb0 * 16 + Nc] = (_Float16)(1.0f - 2.0f / (e0 + 1.0f));
      sH[M * D_HL + nb1 * 16 + Nc] = (_Float16)(1.0f - 2.0f / (e1 + 1.0f));
    }
    __syncthreads();

    // ---- GEMM2 (wave 0): out[16,16] = sigmoid(h @ W2^T + b2) ----
    if (wave == 0) {
      v8f acc2;
      #pragma unroll
      for (int i = 0; i < 8; ++i) acc2[i] = bias2;
      #pragma unroll
      for (int kf = 0; kf < KF2; ++kf) {
        const _Float16* ap = sH + Nc * D_HL + kf * 32 + hiHalf * 16;
        v16h a = cat16(*(const h8*)(ap), *(const h8*)(ap + 8));
        const _Float16* bp = sW2 + Nc * D_HL + kf * 32 + hiHalf * 16;
        v16h bm = cat16(*(const h8*)(bp), *(const h8*)(bp + 8));
        acc2 = wmma_f16(a, bm, acc2);
      }
      const int slot = t & 7;          // overwrite oldest (t-8) with time t
      #pragma unroll
      for (int r = 0; r < 8; ++r) {
        float s = 1.0f / (1.0f + __expf(-acc2[r]));
        int M = mBase + r;
        sHist[((slot * 16 + M) << 4) + Nc] = (_Float16)s;
        out[(((size_t)(b0 + M)) * T_LEN + t) * D_Y + Nc] = s;
      }
    }
    __syncthreads();
  }
}

extern "C" void kernel_launch(void* const* d_in, const int* in_sizes, int n_in,
                              void* d_out, int out_size, void* d_ws, size_t ws_size,
                              hipStream_t stream) {
  const float* x  = (const float*)d_in[0];
  const float* W1 = (const float*)d_in[1];
  const float* b1 = (const float*)d_in[2];
  const float* W2 = (const float*)d_in[3];
  const float* b2 = (const float*)d_in[4];
  float* out = (float*)d_out;

  // LDS: W2(8192) + b1(1024) + b2(64) + inp(12288) + h(8192) + hist(4096)
  const size_t smem = (size_t)(D_Y * D_HL + BTILE * IN_SZ +
                               BTILE * D_HL + D_O * 16 * D_Y) * 2
                    + (size_t)(D_HL + D_Y) * 4;   // = 33856 bytes

  narx_persistent<<<2048 / BTILE, 256, smem, stream>>>(x, W1, b1, W2, b2, out);
}
